// Discriminator_46102178955359
// MI455X (gfx1250) — matrix operand order repaired and checked by reading, not compiled, not run on hardware
//
#include <hip/hip_runtime.h>
#include <hip/hip_bf16.h>
#include <math.h>

// ---------------- problem constants ----------------
#define BATCH  64
#define OO     9          // objects
#define NCLS   5          // classes / layout channels
#define HW     108        // layout spatial (HEIGHT==WIDTH)
#define C1OUT  32
#define C1K    125        // 5*5*5 (padded to 128 in bf16 weight copy)
#define S1     52         // conv1 out spatial
#define C2OUT  64
#define C2K    800        // 32*5*5  (= 25 * 32)
#define S2     24         // conv2 out spatial
#define FC1OUT 512
#define FLATK  36864      // 64*24*24 (= 1152 * 32)
#define EPSV   1e-5f

// fc1 TDM staging geometry
#define FC1_SLICES 8
#define FC1_KSLICE (FLATK / FC1_SLICES)     // 4608
#define FC1_KC     512                      // K floats staged per TDM transfer
#define FC1_STAGES (FC1_KSLICE / FC1_KC)    // 9
#define FC1_LDS_STRIDE 520                  // 512 + TDM pad (4 dw per 256 dw)

// ---------------- WMMA types / helpers ----------------
typedef __attribute__((ext_vector_type(16))) __bf16          v16bf;
typedef __attribute__((ext_vector_type(16))) unsigned short  v16us;
typedef __attribute__((ext_vector_type(8)))  float           v8f;
typedef __attribute__((ext_vector_type(4)))  unsigned int    u32x4;
typedef __attribute__((ext_vector_type(8)))  unsigned int    u32x8;

static __device__ __forceinline__ unsigned short f2bf(float f) {
  // round-to-nearest-even f32 -> bf16 (bit math)
  unsigned int u = __float_as_uint(f);
  u += 0x7FFFu + ((u >> 16) & 1u);
  return (unsigned short)(u >> 16);
}

// pack two f32 into one dword of 2x bf16 (hardware packed convert if available)
#if __has_builtin(__builtin_amdgcn_cvt_pk_bf16_f32)
static __device__ __forceinline__ unsigned pk2(float lo, float hi) {
  return __builtin_bit_cast(unsigned, __builtin_amdgcn_cvt_pk_bf16_f32(lo, hi));
}
#else
static __device__ __forceinline__ unsigned pk2(float lo, float hi) {
  return (unsigned)f2bf(lo) | ((unsigned)f2bf(hi) << 16);
}
#endif

static __device__ __forceinline__ v8f wmma_bf16(v16us a, v16us b, v8f c) {
  // D(16x16 f32) = A(16x32 bf16) * B(32x16 bf16) + C
  return __builtin_amdgcn_wmma_f32_16x16x32_bf16(
      false, __builtin_bit_cast(v16bf, a),
      false, __builtin_bit_cast(v16bf, b),
      (short)0, c, false, false);
}

static __device__ __forceinline__ float leaky(float x) { return x > 0.f ? x : 0.2f * x; }

// compile-time im2col spatial offsets (kb -> ci*plane + kh*row + kw)
constexpr int c1off(int kb) {
  return (kb / 25) * (HW * HW) + ((kb % 25) / 5) * HW + ((kb % 25) % 5);
}
constexpr int c2off(int kb) {
  return (kb / 25) * (S1 * S1) + ((kb % 25) / 5) * S1 + ((kb % 25) % 5);
}

// ---------------- 0) weight prep: f32 -> bf16 (conv1 zero-padded to K=128) ----
__global__ void k_prep_w(const float* __restrict__ w1, const float* __restrict__ w2,
                         unsigned short* __restrict__ w1b, unsigned short* __restrict__ w2b) {
  int i = blockIdx.x * blockDim.x + threadIdx.x;
  if (i < C1OUT * 128) {
    int m = i >> 7, k = i & 127;
    w1b[i] = (k < C1K) ? f2bf(w1[m * C1K + k]) : (unsigned short)0;
  }
  if (i < C2OUT * C2K) w2b[i] = f2bf(w2[i]);
}

// ---------------- 1) layout rasterizer -> bf16 NCHW ----------------
__global__ void k_layout(const float* __restrict__ image, unsigned short* __restrict__ act0) {
  __shared__ float sim[OO * 9];
  int b = blockIdx.y;
  if (threadIdx.x < OO * 9) sim[threadIdx.x] = image[b * OO * 9 + threadIdx.x];
  __syncthreads();
  int pix = blockIdx.x * blockDim.x + threadIdx.x;
  if (pix >= HW * HW) return;
  int y = pix / HW, x = pix % HW;
  float fy = (float)y, fx = (float)x;
  float best[NCLS];
#pragma unroll
  for (int c = 0; c < NCLS; ++c) best[c] = -3.4e38f;
  for (int o = 0; o < OO; ++o) {
    const float* fp = &sim[o * 9];
    float xc = fp[0] * (float)HW, yc = fp[1] * (float)HW;
    float w  = fp[2] * (float)HW, h  = fp[3] * (float)HW;
    float x1 = xc - 0.5f * w, x2 = xc + 0.5f * w;
    float y1 = yc - 0.5f * h, y2 = yc + 0.5f * h;
    float x1d = fx - x1, x2d = x2 - fx, y1d = fy - y1, y2d = y2 - fy;
    float yin = fminf(fmaxf(y1d, 0.f), 1.f) * fminf(fmaxf(y2d, 0.f), 1.f);
    float xin = fminf(fmaxf(x1d, 0.f), 1.f) * fminf(fmaxf(x2d, 0.f), 1.f);
    float l1 = fmaxf(1.f - fabsf(x1d), 0.f) * yin;
    float l2 = fmaxf(1.f - fabsf(x2d), 0.f) * yin;
    float l3 = fmaxf(1.f - fabsf(y1d), 0.f) * xin;
    float l4 = fmaxf(1.f - fabsf(y2d), 0.f) * xin;
    float m  = fmaxf(fmaxf(l1, l2), fmaxf(l3, l4));
#pragma unroll
    for (int c = 0; c < NCLS; ++c) best[c] = fmaxf(best[c], m * fp[4 + c]);
  }
#pragma unroll
  for (int c = 0; c < NCLS; ++c)
    act0[((b * NCLS + c) * HW + y) * HW + x] = f2bf(best[c]);
}

// ---------------- 2) conv1 implicit-im2col WMMA GEMM (bf16 operands) ----------
// M=32 (2 tiles), K=128 padded (4 chunks), N = 64 * 2704 (169 tiles/batch)
__global__ __launch_bounds__(32) void k_conv1(const unsigned short* __restrict__ xbf,
                                              const unsigned short* __restrict__ wbf,
                                              const float* __restrict__ bias,
                                              float* __restrict__ out) {
  int tile = blockIdx.x;
  int mt = tile & 1, nt = tile >> 1;
  int b = nt / 169, pix0 = (nt % 169) * 16;
  int lane = threadIdx.x, half = lane >> 4, l16 = lane & 15;
  int n = pix0 + l16, oh = n / S1, ow = n % S1;
  int m = mt * 16 + l16;
  int base1 = ((b * NCLS) * HW + oh * 2) * HW + ow * 2;  // ci=kh=kw=0 element
  v8f acc = {};
#pragma unroll
  for (int ch = 0; ch < 4; ++ch) {
    u32x8 ta, tb;
    // A: two 8-element bf16 runs -> two b128 loads (guard-free, K padded)
    const uint4* wp = (const uint4*)&wbf[m * 128 + ch * 32 + half * 16];
    uint4 q0 = wp[0], q1 = wp[1];
    ta[0] = q0.x; ta[1] = q0.y; ta[2] = q0.z; ta[3] = q0.w;
    ta[4] = q1.x; ta[5] = q1.y; ta[6] = q1.z; ta[7] = q1.w;
    // B: im2col gather; spatial offsets are compile-time for both half values
#pragma unroll
    for (int v = 0; v < 8; ++v) {
      const int kbL = ch * 32 + v * 2;   // half = 0
      const int kbH = kbL + 16;          // half = 1
      int  o0  = half ? c1off(kbH) : c1off(kbL);
      bool ok0 = half ? (kbH < C1K) : (kbL < C1K);
      unsigned short e0 = ok0 ? xbf[base1 + o0] : (unsigned short)0;
      int  o1  = half ? c1off(kbH + 1) : c1off(kbL + 1);
      bool ok1 = half ? (kbH + 1 < C1K) : (kbL + 1 < C1K);
      unsigned short e1 = ok1 ? xbf[base1 + o1] : (unsigned short)0;
      tb[v] = (unsigned)e0 | ((unsigned)e1 << 16);
    }
    acc = wmma_bf16(__builtin_bit_cast(v16us, ta), __builtin_bit_cast(v16us, tb), acc);
  }
#pragma unroll
  for (int v = 0; v < 8; ++v) {
    int mo = mt * 16 + v + half * 8;
    out[((b * C1OUT + mo) * S1 + oh) * S1 + ow] = acc[v] + bias[mo];
  }
}

// ---------------- per-channel BN stats (mean, 1/sqrt(var+eps)) ----------------
__global__ void k_chan_stats(const float* __restrict__ x, float* __restrict__ stats,
                             int nchan, int per_b, int nb) {
  int c = blockIdx.x;
  int total = nb * per_b;
  float s = 0.f, s2 = 0.f;
  for (int i = threadIdx.x; i < total; i += blockDim.x) {
    int b = i / per_b, p = i % per_b;
    float v = x[(b * nchan + c) * per_b + p];
    s += v; s2 += v * v;
  }
  __shared__ float rs[256], rq[256];
  rs[threadIdx.x] = s; rq[threadIdx.x] = s2;
  __syncthreads();
  for (int off = 128; off > 0; off >>= 1) {
    if ((int)threadIdx.x < off) { rs[threadIdx.x] += rs[threadIdx.x + off]; rq[threadIdx.x] += rq[threadIdx.x + off]; }
    __syncthreads();
  }
  if (threadIdx.x == 0) {
    float mean = rs[0] / (float)total;
    float var  = rq[0] / (float)total - mean * mean;
    stats[c * 2]     = mean;
    stats[c * 2 + 1] = rsqrtf(var + EPSV);
  }
}

// ---------------- BN+leaky apply, emit bf16 activations ----------------
__global__ void k_bn_apply(const float* __restrict__ x, const float* __restrict__ stats,
                           const float* __restrict__ g, const float* __restrict__ bta,
                           unsigned short* __restrict__ out, int nchan, int per_b, int total) {
  int i = blockIdx.x * blockDim.x + threadIdx.x;
  if (i >= total) return;
  int chan = (i / per_b) % nchan;
  float v = (x[i] - stats[chan * 2]) * stats[chan * 2 + 1] * g[chan] + bta[chan];
  out[i] = f2bf(leaky(v));
}

// ---------------- 3) conv2 WMMA GEMM (pure bf16 operands) ----------------
// M=64 (4 tiles), K=800 (25 chunks), N = 64 * 576 (36 tiles/batch)
__global__ __launch_bounds__(32) void k_conv2(const unsigned short* __restrict__ xbf,
                                              const unsigned short* __restrict__ wbf,
                                              const float* __restrict__ bias,
                                              float* __restrict__ out) {
  int tile = blockIdx.x;
  int mt = tile & 3, nt = tile >> 2;
  int b = nt / 36, pix0 = (nt % 36) * 16;
  int lane = threadIdx.x, half = lane >> 4, l16 = lane & 15;
  int n = pix0 + l16, oh = n / S2, ow = n % S2;
  int m = mt * 16 + l16;
  int base2 = ((b * C1OUT) * S1 + oh * 2) * S1 + ow * 2;  // ci=kh=kw=0 element
  v8f acc = {};
#pragma unroll 5
  for (int ch = 0; ch < 25; ++ch) {
    u32x8 ta, tb;
    const uint4* wp = (const uint4*)&wbf[m * C2K + ch * 32 + half * 16];
    uint4 q0 = wp[0], q1 = wp[1];
    ta[0] = q0.x; ta[1] = q0.y; ta[2] = q0.z; ta[3] = q0.w;
    ta[4] = q1.x; ta[5] = q1.y; ta[6] = q1.z; ta[7] = q1.w;
#pragma unroll
    for (int v = 0; v < 8; ++v) {
      const int kbL = ch * 32 + v * 2;   // half = 0
      const int kbH = kbL + 16;          // half = 1
      int o0 = half ? c2off(kbH) : c2off(kbL);
      int o1 = half ? c2off(kbH + 1) : c2off(kbL + 1);
      unsigned short e0 = xbf[base2 + o0];
      unsigned short e1 = xbf[base2 + o1];
      tb[v] = (unsigned)e0 | ((unsigned)e1 << 16);
    }
    acc = wmma_bf16(__builtin_bit_cast(v16us, ta), __builtin_bit_cast(v16us, tb), acc);
  }
#pragma unroll
  for (int v = 0; v < 8; ++v) {
    int mo = mt * 16 + v + half * 8;
    out[((b * C2OUT + mo) * S2 + oh) * S2 + ow] = acc[v] + bias[mo];
  }
}

// ---------------- 4) fc1 WMMA GEMM: TDM-staged f32 weights in LDS ------------
// Grid: 32 M-tiles x 8 K-slices = 256 blocks of 128 threads (4 waves).
// Wave w handles batch columns [16w,16w+16); A tile (16 rows x 512 K) staged by
// the Tensor Data Mover into LDS, shared by all 4 waves (4x HBM reuse).
// Descriptor pads LDS rows (+4 dw / 256 dw) -> stride 520 dwords: bank spread
// with 16B alignment of every 8-float run kept.  B comes from pre-normalized
// bf16 activations: two raw b128 loads, zero VALU.
__global__ __launch_bounds__(128) void k_fc1(const unsigned short* __restrict__ xbf,
                                             const float* __restrict__ wgt,
                                             float* __restrict__ part) {
  __shared__ float smA[16 * FC1_LDS_STRIDE];   // 33,280 B
  int mt    = blockIdx.x >> 3;                 // 0..31
  int slice = blockIdx.x & 7;                  // 0..7
  int wave  = threadIdx.x >> 5;                // 0..3 -> batch column tile
  int lane  = threadIdx.x & 31;
  int half  = lane >> 4, l16 = lane & 15;
  int n = wave * 16 + l16;                     // batch index
  v8f acc = {};

  unsigned long long gbase = (unsigned long long)(uintptr_t)wgt
      + ((unsigned long long)(mt * 16) * FLATK + (unsigned long long)(slice * FC1_KSLICE)) * 4ull;
  unsigned lds_base = (unsigned)(uintptr_t)&smA[0];

  for (int st = 0; st < FC1_STAGES; ++st) {
    __syncthreads();                           // previous stage fully consumed
    if (wave == 0) {
      unsigned long long ga = gbase + (unsigned long long)st * FC1_KC * 4ull;
      // D# group 0: count=1 | lds_addr | global_addr(57b) | type=2
      u32x4 g0 = { 1u,
                   lds_base,
                   (unsigned)ga,
                   (unsigned)((ga >> 32) & 0x01FFFFFFull) | (2u << 30) };
      // D# group 1: data_size=4B(code2), pad_enable, pad_interval=256dw(code7),
      //             pad_amount=4dw(code3); tensor_dim0=36864, tensor_dim1=512,
      //             tile_dim0=512, tile_dim1=16, dim0_stride=36864
      u32x8 g1 = { (2u << 16) | (1u << 20) | (7u << 22) | (3u << 25),
                   ((unsigned)(FLATK & 0xFFFF)) << 16,
                   ((unsigned)(FLATK >> 16) & 0xFFFFu) | (512u << 16),
                   ((unsigned)FC1_KC) << 16,
                   16u,
                   (unsigned)FLATK,
                   0u, 0u };
      u32x4 gz = { 0u, 0u, 0u, 0u };           // groups 2/3: unused dims = 0
      asm volatile("tensor_load_to_lds %0, %1, %2, %3"
                   :: "s"(g0), "s"(g1), "s"(gz), "s"(gz)
                   : "memory");
      __builtin_amdgcn_s_wait_tensorcnt(0);
    }
    __syncthreads();                           // staged tile visible to all waves

    int kb_glob0 = slice * FC1_KSLICE + st * FC1_KC;
    const int row = l16 * FC1_LDS_STRIDE;
#pragma unroll 4
    for (int cc = 0; cc < FC1_KC / 32; ++cc) {
      u32x8 ta, tb;
      // ---- A from LDS f32 (pad-aware), packed convert to bf16 ----
      {
        int b0 = cc * 32 + half * 16;
        int b1 = b0 + 8;
        int o0 = row + b0 + 4 * (b0 >> 8);
        int o1 = row + b1 + 4 * (b1 >> 8);
        float4 a0 = *(const float4*)&smA[o0];
        float4 a1 = *(const float4*)&smA[o0 + 4];
        float4 a2 = *(const float4*)&smA[o1];
        float4 a3 = *(const float4*)&smA[o1 + 4];
        ta[0] = pk2(a0.x, a0.y); ta[1] = pk2(a0.z, a0.w);
        ta[2] = pk2(a1.x, a1.y); ta[3] = pk2(a1.z, a1.w);
        ta[4] = pk2(a2.x, a2.y); ta[5] = pk2(a2.z, a2.w);
        ta[6] = pk2(a3.x, a3.y); ta[7] = pk2(a3.z, a3.w);
      }
      // ---- B: 16 contiguous pre-normalized bf16 -> two b128 loads ----
      {
        int kb = kb_glob0 + cc * 32 + half * 16;
        const uint4* xp = (const uint4*)&xbf[(size_t)n * FLATK + kb];
        uint4 b0 = xp[0], b1 = xp[1];
        tb[0] = b0.x; tb[1] = b0.y; tb[2] = b0.z; tb[3] = b0.w;
        tb[4] = b1.x; tb[5] = b1.y; tb[6] = b1.z; tb[7] = b1.w;
      }
      acc = wmma_bf16(__builtin_bit_cast(v16us, ta), __builtin_bit_cast(v16us, tb), acc);
    }
  }
#pragma unroll
  for (int v = 0; v < 8; ++v) {
    int mo = mt * 16 + v + half * 8;
    part[(slice * BATCH + n) * FC1OUT + mo] = acc[v];
  }
}

__global__ void k_fc1_reduce(const float* __restrict__ part, const float* __restrict__ bias,
                             float* __restrict__ out) {
  int i = blockIdx.x * blockDim.x + threadIdx.x;   // over 64*512
  if (i >= BATCH * FC1OUT) return;
  int m = i % FC1OUT;
  float s = bias[m];
#pragma unroll
  for (int sl = 0; sl < FC1_SLICES; ++sl) s += part[sl * BATCH * FC1OUT + i];
  out[i] = s;
}

// ---------------- 5) BN3 stats (per feature, over batch axis) ----------------
__global__ void k_feat_stats(const float* __restrict__ x, float* __restrict__ stats) {
  int f = blockIdx.x * blockDim.x + threadIdx.x;
  if (f >= FC1OUT) return;
  float s = 0.f, s2 = 0.f;
  for (int b = 0; b < BATCH; ++b) { float v = x[b * FC1OUT + f]; s += v; s2 += v * v; }
  float mean = s / (float)BATCH;
  float var  = s2 / (float)BATCH - mean * mean;
  stats[f * 2]     = mean;
  stats[f * 2 + 1] = rsqrtf(var + EPSV);
}

// ---------------- 6) fc2 + BN3 + leaky + sigmoid ----------------
__global__ void k_fc2(const float* __restrict__ fc1o, const float* __restrict__ stats3,
                      const float* __restrict__ g3, const float* __restrict__ b3,
                      const float* __restrict__ w2, const float* __restrict__ bb2,
                      float* __restrict__ out) {
  int b = blockIdx.x;
  float s = 0.f;
  for (int f = threadIdx.x; f < FC1OUT; f += blockDim.x) {
    float raw = fc1o[b * FC1OUT + f];
    float a = leaky((raw - stats3[f * 2]) * stats3[f * 2 + 1] * g3[f] + b3[f]);
    s += a * w2[f];
  }
  __shared__ float rs[128];
  rs[threadIdx.x] = s;
  __syncthreads();
  for (int off = 64; off > 0; off >>= 1) {
    if ((int)threadIdx.x < off) rs[threadIdx.x] += rs[threadIdx.x + off];
    __syncthreads();
  }
  if (threadIdx.x == 0) {
    float z = rs[0] + bb2[0];
    out[b] = 1.f / (1.f + expf(-z));
  }
}

// ---------------- host-side launch ----------------
extern "C" void kernel_launch(void* const* d_in, const int* in_sizes, int n_in,
                              void* d_out, int out_size, void* d_ws, size_t ws_size,
                              hipStream_t stream) {
  (void)in_sizes; (void)n_in; (void)out_size; (void)ws_size;
  const float* image   = (const float*)d_in[0];
  const float* conv1_w = (const float*)d_in[1];
  const float* conv1_b = (const float*)d_in[2];
  const float* bn1_g   = (const float*)d_in[3];
  const float* bn1_b   = (const float*)d_in[4];
  const float* conv2_w = (const float*)d_in[5];
  const float* conv2_b = (const float*)d_in[6];
  const float* bn2_g   = (const float*)d_in[7];
  const float* bn2_b   = (const float*)d_in[8];
  const float* fc1_w   = (const float*)d_in[9];
  const float* fc1_b   = (const float*)d_in[10];
  const float* bn3_g   = (const float*)d_in[11];
  const float* bn3_b   = (const float*)d_in[12];
  const float* fc2_w   = (const float*)d_in[13];
  const float* fc2_b   = (const float*)d_in[14];
  float* out = (float*)d_out;

  // workspace partition; ~56.1 MB total, all sub-buffers 16B aligned
  float* c1o = (float*)d_ws;                         // 64*32*52*52 = 5,537,792 f32
  float* c2o = c1o + 5537792;                        // 64*64*24*24 = 2,359,296 f32
  float* f1o = c2o + 2359296;                        // 64*512      =    32,768 f32
  float* s1  = f1o + 32768;                          // 32*2
  float* s2  = s1  + 64;                             // 64*2
  float* s3  = s2  + 128;                            // 512*2
  float* f1p = s3  + 1024;                           // 8*64*512    =   262,144 f32
  unsigned short* act0bf = (unsigned short*)(f1p + 262144);  // 3,732,480 bf16
  unsigned short* act1bf = act0bf + 3732480;                 // 5,537,792 bf16
  unsigned short* act2bf = act1bf + 5537792;                 // 2,359,296 bf16
  unsigned short* w1b    = act2bf + 2359296;                 // 32*128 bf16 (padded)
  unsigned short* w2b    = w1b + 4096;                       // 64*800 bf16

  k_prep_w<<<200, 256, 0, stream>>>(conv1_w, conv2_w, w1b, w2b);
  k_layout<<<dim3(46, 64), 256, 0, stream>>>(image, act0bf);
  k_conv1<<<64 * 169 * 2, 32, 0, stream>>>(act0bf, w1b, conv1_b, c1o);
  k_chan_stats<<<C1OUT, 256, 0, stream>>>(c1o, s1, C1OUT, S1 * S1, BATCH);
  k_bn_apply<<<(5537792 + 255) / 256, 256, 0, stream>>>(c1o, s1, bn1_g, bn1_b,
                                                        act1bf, C1OUT, S1 * S1, 5537792);
  k_conv2<<<64 * 36 * 4, 32, 0, stream>>>(act1bf, w2b, conv2_b, c2o);
  k_chan_stats<<<C2OUT, 256, 0, stream>>>(c2o, s2, C2OUT, S2 * S2, BATCH);
  k_bn_apply<<<(2359296 + 255) / 256, 256, 0, stream>>>(c2o, s2, bn2_g, bn2_b,
                                                        act2bf, C2OUT, S2 * S2, 2359296);
  k_fc1<<<32 * FC1_SLICES, 128, 0, stream>>>(act2bf, fc1_w, f1p);
  k_fc1_reduce<<<(BATCH * FC1OUT + 255) / 256, 256, 0, stream>>>(f1p, fc1_b, f1o);
  k_feat_stats<<<2, 256, 0, stream>>>(f1o, s3);
  k_fc2<<<BATCH, 128, 0, stream>>>(f1o, s3, bn3_g, bn3_b, fc2_w, fc2_b, out);
}
